// Attention_25048249270293
// MI455X (gfx1250) — matrix operand order repaired and checked by reading, compiled, not run on hardware
//
#include <hip/hip_runtime.h>
#include <hip/hip_bf16.h>

typedef __attribute__((ext_vector_type(16))) _Float16 v16h;
typedef __attribute__((ext_vector_type(8)))  _Float16 v8h;
typedef __attribute__((ext_vector_type(2)))  _Float16 v2h;
typedef __attribute__((ext_vector_type(8)))  float    v8f;
typedef __attribute__((ext_vector_type(4)))  float    v4f;

#define LQ_TILE 128   // q rows per workgroup (8 waves x 16)
#define KB      32    // keys per main-loop iteration
#define DM      128   // head dim

// LDS tiles (f16), strides padded but kept 16B-aligned:
//   K  tile row stride: 136 halves = 272 B (17*16)
//   Vt tile row stride:  40 halves =  80 B ( 5*16)
#define KL_STRIDE  (DM + 8)
#define VT_STRIDE  (KB + 8)

__global__ __launch_bounds__(256)
void fa_fwd_f16wmma(const float* __restrict__ Q, const float* __restrict__ K,
                    const float* __restrict__ V, float* __restrict__ O,
                    int Lq, int Lk)
{
    // double-buffered tiles for software pipelining
    __shared__ _Float16 Kl[2][KB][KL_STRIDE];   // [buf][k][d]
    __shared__ _Float16 Vt[2][DM][VT_STRIDE];   // [buf][d][k] (V transposed)

    const int tid  = threadIdx.x;
    const int lane = tid & 31;
    const int wave = tid >> 5;
    const int lq   = lane & 15;
    const bool hi  = lane >= 16;

    const int batch = blockIdx.y;
    const int qbase = blockIdx.x * LQ_TILE + wave * 16;

    const float scale = 0.08838834764831845f;   // 1/sqrt(128)
    const float log2e = 1.44269504088896340f;

    const float* Qb = Q + ((size_t)batch * Lq + qbase + lq) * DM;
    const float* Kb = K + (size_t)batch * Lk * DM;
    const float* Vb = V + (size_t)batch * Lk * DM;

    // ---- Q fragments: B-operand of S^T = K * Q^T, held for the whole loop.
    v16h qf[4];
    #pragma unroll
    for (int c = 0; c < 4; ++c) {
        const int dstart = c * 32 + (hi ? 16 : 0);
        const v4f* src = (const v4f*)(Qb + dstart);
        #pragma unroll
        for (int j4 = 0; j4 < 4; ++j4) {
            v4f x = src[j4];
            #pragma unroll
            for (int e = 0; e < 4; ++e)
                qf[c][j4 * 4 + e] = (_Float16)(x[e] * scale);
        }
    }

    // O^T accumulators: 8 d-tiles of 16x16 f32 (lane = q, rows = d)
    v8f oacc[8];
    #pragma unroll
    for (int i = 0; i < 8; ++i)
        #pragma unroll
        for (int r = 0; r < 8; ++r) oacc[i][r] = 0.0f;

    float mrun = -3.0e38f;
    float lrun = 0.0f;

    // ---- loader-role indices (per thread, fixed) ----
    const int krow  = tid >> 3;          // 0..31  (K tile row)
    const int kcol0 = (tid & 7) * 16;    // 0..112 (K tile col start)
    const int vk0   = (tid & 15) * 2;    // 0..30  (V keys pair)
    const int vd0   = (tid >> 4) * 8;    // 0..120 (V dims chunk)

    float kreg[16], va0[8], va1[8];      // in-flight tile registers

    auto load_tile = [&](int kstart) {
        const v4f* src = (const v4f*)(Kb + (size_t)(kstart + krow) * DM + kcol0);
        #pragma unroll
        for (int j4 = 0; j4 < 4; ++j4) {
            v4f x = src[j4];
            #pragma unroll
            for (int e = 0; e < 4; ++e) kreg[j4 * 4 + e] = x[e];
        }
        const v4f* s0 = (const v4f*)(Vb + (size_t)(kstart + vk0) * DM + vd0);
        const v4f* s1 = (const v4f*)(Vb + (size_t)(kstart + vk0 + 1) * DM + vd0);
        #pragma unroll
        for (int j4 = 0; j4 < 2; ++j4) {
            v4f x0 = s0[j4], x1 = s1[j4];
            #pragma unroll
            for (int e = 0; e < 4; ++e) {
                va0[j4 * 4 + e] = x0[e];
                va1[j4 * 4 + e] = x1[e];
            }
        }
        // prefetch one more tile ahead into L2
        __builtin_prefetch((const char*)(src) + (size_t)KB * DM * 4, 0, 1);
    };

    auto store_tile = [&](int buf) {
        v8h lo8, hi8;
        #pragma unroll
        for (int e = 0; e < 8; ++e) {
            lo8[e] = (_Float16)kreg[e];
            hi8[e] = (_Float16)kreg[8 + e];
        }
        *(v8h*)&Kl[buf][krow][kcol0]     = lo8;
        *(v8h*)&Kl[buf][krow][kcol0 + 8] = hi8;
        #pragma unroll
        for (int j = 0; j < 8; ++j) {
            v2h p;
            p[0] = (_Float16)va0[j];
            p[1] = (_Float16)va1[j];
            *(v2h*)&Vt[buf][vd0 + j][vk0] = p;
        }
    };

    const int kIters = Lk / KB;

    // ---- pipeline prologue: tile 0 into buffer 0 ----
    load_tile(0);
    store_tile(0);
    __syncthreads();

    for (int kb = 0; kb < kIters; ++kb) {
        const int cur = kb & 1;
        const bool more = (kb + 1) < kIters;

        // issue next tile's global loads early; latency hidden by WMMAs below
        if (more) load_tile((kb + 1) * KB);

        // ---- S^T = K_tile * Q^T : two 16(k) x 16(q) tiles, K-dim = d = 128
        v8f st[2];
        #pragma unroll
        for (int t = 0; t < 2; ++t) {
            #pragma unroll
            for (int r = 0; r < 8; ++r) st[t][r] = 0.0f;
            const int kr = t * 16 + lq;
            // preload all 4 A fragments of this tile, then chain WMMAs
            v16h a[4];
            #pragma unroll
            for (int c = 0; c < 4; ++c) {
                const int base = c * 32 + (hi ? 16 : 0);
                v8h a0 = *(const v8h*)&Kl[cur][kr][base];
                v8h a1 = *(const v8h*)&Kl[cur][kr][base + 8];
                #pragma unroll
                for (int e = 0; e < 8; ++e) { a[c][e] = a0[e]; a[c][8 + e] = a1[e]; }
            }
            #pragma unroll
            for (int c = 0; c < 4; ++c)
                st[t] = __builtin_amdgcn_wmma_f32_16x16x32_f16(
                            false, a[c], false, qf[c], (short)0, st[t], false, false);
        }

        // ---- online softmax over k (rows): per-lane + one xor-16 merge
        float smax = st[0][0];
        #pragma unroll
        for (int r = 1; r < 8; ++r) smax = fmaxf(smax, st[0][r]);
        #pragma unroll
        for (int r = 0; r < 8; ++r) smax = fmaxf(smax, st[1][r]);
        smax = fmaxf(smax, __shfl_xor(smax, 16, 32));
        const float mnew  = fmaxf(mrun, smax);
        const float alpha = exp2f((mrun - mnew) * log2e);

        float p[16];
        float psum = 0.0f;
        #pragma unroll
        for (int t = 0; t < 2; ++t)
            #pragma unroll
            for (int r = 0; r < 8; ++r) {
                float e = exp2f((st[t][r] - mnew) * log2e);
                p[t * 8 + r] = e;
                psum += e;
            }
        psum += __shfl_xor(psum, 16, 32);
        lrun = lrun * alpha + psum;
        mrun = mnew;

        #pragma unroll
        for (int i = 0; i < 8; ++i)
            #pragma unroll
            for (int r = 0; r < 8; ++r) oacc[i][r] *= alpha;

        // ---- build P^T as the B operand (32k x 16q): half-wave exchange
        v16h pf;
        #pragma unroll
        for (int j = 0; j < 8; ++j) {
            float send = hi ? p[j] : p[8 + j];
            float recv = __shfl_xor(send, 16, 32);
            float lo_v = hi ? recv     : p[j];
            float hi_v = hi ? p[8 + j] : recv;
            pf[j]     = (_Float16)lo_v;
            pf[8 + j] = (_Float16)hi_v;
        }

        // ---- O^T += V^T * P^T : 8 independent d-tile WMMAs, K-dim = 32 keys
        // ping-pong the A fragment so loads overlap WMMA issue
        const int kh = hi ? 16 : 0;
        v16h av[2];
        {
            v8h a0 = *(const v8h*)&Vt[cur][lq][kh];
            v8h a1 = *(const v8h*)&Vt[cur][lq][kh + 8];
            #pragma unroll
            for (int e = 0; e < 8; ++e) { av[0][e] = a0[e]; av[0][8 + e] = a1[e]; }
        }
        #pragma unroll
        for (int dt = 0; dt < 8; ++dt) {
            if (dt < 7) {
                const int drow = (dt + 1) * 16 + lq;
                v8h a0 = *(const v8h*)&Vt[cur][drow][kh];
                v8h a1 = *(const v8h*)&Vt[cur][drow][kh + 8];
                #pragma unroll
                for (int e = 0; e < 8; ++e) {
                    av[(dt + 1) & 1][e]     = a0[e];
                    av[(dt + 1) & 1][8 + e] = a1[e];
                }
            }
            oacc[dt] = __builtin_amdgcn_wmma_f32_16x16x32_f16(
                           false, av[dt & 1], false, pf, (short)0, oacc[dt],
                           false, false);
        }

        // ---- pipeline epilogue: convert + store next tile, single barrier
        if (more) {
            store_tile(1 - cur);
            __syncthreads();
        }
    }

    // ---- epilogue: normalize and scatter O^T back as O[q][d]
    const float rinv = 1.0f / lrun;
    float* Ob = O + ((size_t)batch * Lq + qbase + lq) * DM;
    #pragma unroll
    for (int dt = 0; dt < 8; ++dt)
        #pragma unroll
        for (int r = 0; r < 8; ++r) {
            const int d = dt * 16 + r + (hi ? 8 : 0);
            Ob[d] = oacc[dt][r] * rinv;
        }
}

extern "C" void kernel_launch(void* const* d_in, const int* in_sizes, int n_in,
                              void* d_out, int out_size, void* d_ws, size_t ws_size,
                              hipStream_t stream) {
    const float* Q = (const float*)d_in[0];
    const float* K = (const float*)d_in[1];
    const float* V = (const float*)d_in[2];
    float* O = (float*)d_out;

    const int B  = 8;
    const int Lq = in_sizes[0] / (B * DM);
    const int Lk = in_sizes[1] / (B * DM);

    dim3 grid(Lq / LQ_TILE, B);
    fa_fwd_f16wmma<<<grid, 256, 0, stream>>>(Q, K, V, O, Lq, Lk);
}
